// Attention_75453985457143
// MI455X (gfx1250) — matrix operand order repaired and checked by reading, not compiled, not run on hardware
//
#include <hip/hip_runtime.h>
#include <hip/hip_bf16.h>
#include <math.h>

#define RESN   196
#define NPAD   224           // 196 padded to 14 * 16
#define NPT    14            // NPAD / 16
#define DIMC   384
#define NHEADS 12
#define KD     32
#define DVC    128
#define NH_KD  384
#define DH     1536
#define HQKV   2304
#define BATCH  128
#define BN_EPS 1e-5f

typedef _Float16 v16h __attribute__((ext_vector_type(16)));
typedef float    v8f  __attribute__((ext_vector_type(8)));

__device__ __forceinline__ v8f wmma16(v16h a, v16h b, v8f c) {
  // D(f32 16x16) = A(f16 16x32) * B(f16 32x16) + C
  return __builtin_amdgcn_wmma_f32_16x16x32_f16(false, a, false, b, (short)0, c,
                                                false, false);
}

// A-fragment (16x32 f16) per-lane load, f32 source, row-major with K contiguous.
// Lane holds M = lane&15; low half-wave K in {k0..k0+15},
// high half-wave K in {k0+16..k0+31}  (contiguous, matches B loads).
__device__ __forceinline__ v16h a_frag_f32(const float* row, int k0) {
  const int half = (threadIdx.x & 31) >> 4;
  const float* p0 = row + k0 + half * 16;
  v16h a;
#pragma unroll
  for (int j = 0; j < 8; ++j) {
    a[j]     = (_Float16)p0[j];
    a[8 + j] = (_Float16)p0[8 + j];
  }
  return a;
}

__device__ __forceinline__ v16h a_frag_f16(const _Float16* row, int k0) {
  const int half = (threadIdx.x & 31) >> 4;
  const _Float16* p0 = row + k0 + half * 16;
  v16h a;
#pragma unroll
  for (int j = 0; j < 8; ++j) {
    a[j]     = p0[j];
    a[8 + j] = p0[8 + j];
  }
  return a;
}

// ---------------------------------------------------------------------------
// Kernel 1: QKV GEMM  C[o,p] = sum_c W[o,c] * X[b,c,p]  + folded BN.
// Block = 128 threads (4 waves); wave w owns output rows [obase+16w, +16)
// over all 14 padded N-tiles. X tile staged f16 in LDS as [p][c].
// ---------------------------------------------------------------------------
__global__ void __launch_bounds__(128)
qkv_gemm_kernel(const float* __restrict__ x, const float* __restrict__ qkv_w,
                const float* __restrict__ g, const float* __restrict__ bb,
                const float* __restrict__ mm, const float* __restrict__ vv,
                float* __restrict__ qpre, _Float16* __restrict__ kbuf,
                _Float16* __restrict__ vbuf) {
  __shared__ __align__(32) _Float16 Bs[32 * NPAD];  // [p][c]
  const int b = blockIdx.y;
  const int obase = blockIdx.x * 64;
  const int tid = threadIdx.x;
  const int wave = tid >> 5, lane = tid & 31;
  const int lo = lane & 15, half = lane >> 4;
  const int orow = obase + wave * 16;

  v8f acc[NPT];
#pragma unroll
  for (int t = 0; t < NPT; ++t) acc[t] = v8f{};

  const float* xb = x + (size_t)b * DIMC * RESN;
  for (int kk = 0; kk < DIMC / 32; ++kk) {
    const int c0 = kk * 32;
    for (int i = tid; i < 32 * NPAD; i += 128) {
      const int c = i / NPAD, p = i % NPAD;
      const float xv = (p < RESN) ? xb[(size_t)(c0 + c) * RESN + p] : 0.f;
      Bs[p * 32 + c] = (_Float16)xv;
    }
    __syncthreads();
    const v16h a = a_frag_f32(qkv_w + (size_t)(orow + lo) * DIMC, c0);
#pragma unroll
    for (int t = 0; t < NPT; ++t) {
      const v16h bm = *(const v16h*)&Bs[(t * 16 + lo) * 32 + half * 16];
      acc[t] = wmma16(a, bm, acc[t]);
    }
    __syncthreads();
  }

  // epilogue: BN fold + scatter into q (fp32), k (f16 [m][d]), v (f16 [d][m])
#pragma unroll
  for (int r = 0; r < 8; ++r) {
    const int o = orow + r + half * 8;
    const float sc = g[o] * rsqrtf(vv[o] + BN_EPS);
    const float sh = bb[o] - mm[o] * sc;
#pragma unroll
    for (int t = 0; t < NPT; ++t) {
      const int p = t * 16 + lo;
      const float val = acc[t][r] * sc + sh;
      if (o < NH_KD) {
        if (p < RESN) qpre[((size_t)b * NH_KD + o) * RESN + p] = val;
      } else if (o < 2 * NH_KD) {
        const int oc = o - NH_KD, h = oc >> 5, d = oc & 31;
        kbuf[(((size_t)b * NHEADS + h) * NPAD + p) * KD + d] = (_Float16)val;
      } else {
        const int oc = o - 2 * NH_KD, h = oc >> 7, dv = oc & 127;
        vbuf[(((size_t)b * NHEADS + h) * DVC + dv) * NPAD + p] = (_Float16)val;
      }
    }
  }
}

// ---------------------------------------------------------------------------
// Kernel 2: depthwise 3x3 conv on q + BN; emits q in f16 [b][h][n][32] layout.
// ---------------------------------------------------------------------------
__global__ void __launch_bounds__(256)
dwconv_kernel(const float* __restrict__ qpre, const float* __restrict__ dw_w,
              const float* __restrict__ g, const float* __restrict__ bb,
              const float* __restrict__ mm, const float* __restrict__ vv,
              _Float16* __restrict__ qbuf) {
  const int idx = blockIdx.x * 256 + threadIdx.x;
  if (idx >= BATCH * NH_KD * NPAD) return;
  const int n = idx % NPAD;
  const int c = (idx / NPAD) % NH_KD;
  const int b = idx / (NPAD * NH_KD);
  float val = 0.f;
  if (n < RESN) {
    const int y = n / 14, xp = n % 14;
    const float* w9 = dw_w + c * 9;
    const float* src = qpre + ((size_t)b * NH_KD + c) * RESN;
    float s = 0.f;
#pragma unroll
    for (int ky = 0; ky < 3; ++ky) {
      const int yy = y + ky - 1;
      if (yy < 0 || yy >= 14) continue;
#pragma unroll
      for (int kx = 0; kx < 3; ++kx) {
        const int xx = xp + kx - 1;
        if (xx < 0 || xx >= 14) continue;
        s += w9[ky * 3 + kx] * src[yy * 14 + xx];
      }
    }
    const float sc = g[c] * rsqrtf(vv[c] + BN_EPS);
    val = s * sc + (bb[c] - mm[c] * sc);
  }
  const int h = c >> 5, d = c & 31;
  qbuf[(((size_t)b * NHEADS + h) * NPAD + n) * KD + d] = (_Float16)val;
}

// ---------------------------------------------------------------------------
// Kernel 3: fused attention per (b,h). Each wave owns a 16-row strip of S:
//   S = q^T k  (one K=32 WMMA per 16x16 tile), bias + mask, in-register
//   softmax (shfl_xor width-16 butterflies), P strip -> private LDS (f16),
//   O_strip = v * P^T via WMMA, ReLU fused at the store.
// ---------------------------------------------------------------------------
__global__ void __launch_bounds__(256)
attn_kernel(const _Float16* __restrict__ qbuf, const _Float16* __restrict__ kbuf,
            const _Float16* __restrict__ vbuf, const float* __restrict__ biases,
            const int* __restrict__ bidx, _Float16* __restrict__ obuf) {
  __shared__ __align__(32) _Float16 Pst[8][16 * NPAD];  // 57344 B
  const int bh = blockIdx.x;
  const int b = bh / NHEADS, h = bh % NHEADS;
  const int wave = threadIdx.x >> 5, lane = threadIdx.x & 31;
  const int lo = lane & 15, half = lane >> 4;
  const _Float16* qh = qbuf + (size_t)bh * NPAD * KD;   // [n][32]
  const _Float16* kh = kbuf + (size_t)bh * NPAD * KD;   // [m][32]
  const _Float16* vh = vbuf + (size_t)bh * DVC * NPAD;  // [d][m]
  const float* biash = biases + h * RESN;
  const float scale = 0.1767766952966369f;  // 32^-0.5

  for (int s = wave; s < 13; s += 8) {  // 13 strips cover 208 >= 196 rows
    // -- phase 1: score strip (16 x NPAD)
    const v16h a = a_frag_f16(qh + (s * 16 + lo) * KD, 0);
    v8f S[NPT];
#pragma unroll
    for (int t = 0; t < NPT; ++t) {
      const v16h bm = *(const v16h*)&kh[(t * 16 + lo) * KD + half * 16];
      v8f z{};
      S[t] = wmma16(a, bm, z);
    }
    // -- scale + relative position bias + OOB mask, track row max
    float rmax[8];
#pragma unroll
    for (int r = 0; r < 8; ++r) rmax[r] = -3.0e38f;
#pragma unroll
    for (int r = 0; r < 8; ++r) {
      const int n = s * 16 + r + half * 8;
#pragma unroll
      for (int t = 0; t < NPT; ++t) {
        const int m = t * 16 + lo;
        float sv = -1.0e30f;
        if (n < RESN && m < RESN)
          sv = S[t][r] * scale + biash[bidx[n * RESN + m]];
        S[t][r] = sv;
        rmax[r] = fmaxf(rmax[r], sv);
      }
    }
#pragma unroll
    for (int r = 0; r < 8; ++r)
#pragma unroll
      for (int d = 8; d >= 1; d >>= 1)
        rmax[r] = fmaxf(rmax[r], __shfl_xor(rmax[r], d, 16));
    float rsum[8];
#pragma unroll
    for (int r = 0; r < 8; ++r) rsum[r] = 0.f;
#pragma unroll
    for (int r = 0; r < 8; ++r)
#pragma unroll
      for (int t = 0; t < NPT; ++t) {
        const float e = __expf(S[t][r] - rmax[r]);
        S[t][r] = e;
        rsum[r] += e;
      }
#pragma unroll
    for (int r = 0; r < 8; ++r) {
#pragma unroll
      for (int d = 8; d >= 1; d >>= 1)
        rsum[r] += __shfl_xor(rsum[r], d, 16);
      rsum[r] = 1.f / rsum[r];
    }
    // -- normalized P strip to this wave's private LDS region (f16 [n_l][m])
#pragma unroll
    for (int r = 0; r < 8; ++r) {
      const int nl = r + half * 8;
#pragma unroll
      for (int t = 0; t < NPT; ++t)
        Pst[wave][nl * NPAD + t * 16 + lo] = (_Float16)(S[t][r] * rsum[r]);
    }
    // -- phase 2: O_strip(128 x 16) = v(128 x NPAD) * P_strip^T, K = NPAD
    for (int dt = 0; dt < DVC / 16; ++dt) {
      v8f oacc{};
#pragma unroll
      for (int kk = 0; kk < NPAD / 32; ++kk) {
        const v16h av = a_frag_f16(vh + (dt * 16 + lo) * NPAD, kk * 32);
        const v16h bm =
            *(const v16h*)&Pst[wave][lo * NPAD + kk * 32 + half * 16];
        oacc = wmma16(av, bm, oacc);
      }
      const int n = s * 16 + lo;
      if (n < RESN) {
#pragma unroll
        for (int r = 0; r < 8; ++r) {
          const int dvv = dt * 16 + r + half * 8;
          obuf[((size_t)b * DH + h * DVC + dvv) * NPAD + n] =
              (_Float16)fmaxf(oacc[r], 0.f);  // fused ReLU
        }
      }
    }
  }
}

// ---------------------------------------------------------------------------
// Kernel 4: projection GEMM  out[o,p] = sum_d Wp[o,d] * O[b,d,p]  + BN.
// ---------------------------------------------------------------------------
__global__ void __launch_bounds__(128)
proj_gemm_kernel(const _Float16* __restrict__ obuf,
                 const float* __restrict__ proj_w, const float* __restrict__ g,
                 const float* __restrict__ bb, const float* __restrict__ mm,
                 const float* __restrict__ vv, float* __restrict__ out) {
  __shared__ __align__(32) _Float16 Bs[32 * NPAD];  // [p][d]
  const int b = blockIdx.y;
  const int obase = blockIdx.x * 64;
  const int tid = threadIdx.x;
  const int wave = tid >> 5, lane = tid & 31;
  const int lo = lane & 15, half = lane >> 4;
  const int orow = obase + wave * 16;

  v8f acc[NPT];
#pragma unroll
  for (int t = 0; t < NPT; ++t) acc[t] = v8f{};

  const _Float16* ob = obuf + (size_t)b * DH * NPAD;
  for (int kk = 0; kk < DH / 32; ++kk) {
    const int d0 = kk * 32;
    for (int i = tid; i < 32 * NPAD; i += 128) {
      const int d = i / NPAD, p = i % NPAD;
      _Float16 v = (_Float16)0.f;
      if (p < RESN) v = ob[(size_t)(d0 + d) * NPAD + p];
      Bs[p * 32 + d] = v;
    }
    __syncthreads();
    const v16h a = a_frag_f32(proj_w + (size_t)(orow + lo) * DH, d0);
#pragma unroll
    for (int t = 0; t < NPT; ++t) {
      const v16h bm = *(const v16h*)&Bs[(t * 16 + lo) * 32 + half * 16];
      acc[t] = wmma16(a, bm, acc[t]);
    }
    __syncthreads();
  }
#pragma unroll
  for (int r = 0; r < 8; ++r) {
    const int o = orow + r + half * 8;
    const float sc = g[o] * rsqrtf(vv[o] + BN_EPS);
    const float sh = bb[o] - mm[o] * sc;
#pragma unroll
    for (int t = 0; t < NPT; ++t) {
      const int p = t * 16 + lo;
      if (p < RESN) out[((size_t)b * DIMC + o) * RESN + p] = acc[t][r] * sc + sh;
    }
  }
}

// ---------------------------------------------------------------------------
static inline size_t align256(size_t v) { return (v + 255) & ~(size_t)255; }

extern "C" void kernel_launch(void* const* d_in, const int* in_sizes, int n_in,
                              void* d_out, int out_size, void* d_ws,
                              size_t ws_size, hipStream_t stream) {
  const float* x       = (const float*)d_in[0];
  const float* qkv_w   = (const float*)d_in[1];
  const float* qkv_g   = (const float*)d_in[2];
  const float* qkv_b   = (const float*)d_in[3];
  const float* qkv_m   = (const float*)d_in[4];
  const float* qkv_v   = (const float*)d_in[5];
  const float* dw_w    = (const float*)d_in[6];
  const float* dw_g    = (const float*)d_in[7];
  const float* dw_b    = (const float*)d_in[8];
  const float* dw_m    = (const float*)d_in[9];
  const float* dw_v    = (const float*)d_in[10];
  const float* proj_w  = (const float*)d_in[11];
  const float* proj_g  = (const float*)d_in[12];
  const float* proj_b  = (const float*)d_in[13];
  const float* proj_m  = (const float*)d_in[14];
  const float* proj_v  = (const float*)d_in[15];
  const float* biases  = (const float*)d_in[16];
  const int*   bidx    = (const int*)d_in[17];
  float* out = (float*)d_out;

  char* ws = (char*)d_ws;
  float* qpre = (float*)ws;
  ws += align256((size_t)BATCH * NH_KD * RESN * sizeof(float));
  _Float16* qbuf = (_Float16*)ws;
  ws += align256((size_t)BATCH * NHEADS * NPAD * KD * sizeof(_Float16));
  _Float16* kbuf = (_Float16*)ws;
  ws += align256((size_t)BATCH * NHEADS * NPAD * KD * sizeof(_Float16));
  _Float16* vbuf = (_Float16*)ws;
  ws += align256((size_t)BATCH * NHEADS * DVC * NPAD * sizeof(_Float16));
  _Float16* obuf = (_Float16*)ws;
  ws += align256((size_t)BATCH * DH * NPAD * sizeof(_Float16));

  qkv_gemm_kernel<<<dim3(HQKV / 64, BATCH), 128, 0, stream>>>(
      x, qkv_w, qkv_g, qkv_b, qkv_m, qkv_v, qpre, kbuf, vbuf);

  dwconv_kernel<<<(BATCH * NH_KD * NPAD) / 256, 256, 0, stream>>>(
      qpre, dw_w, dw_g, dw_b, dw_m, dw_v, qbuf);

  attn_kernel<<<BATCH * NHEADS, 256, 0, stream>>>(qbuf, kbuf, vbuf, biases,
                                                  bidx, obuf);

  proj_gemm_kernel<<<dim3(DIMC / 64, BATCH), 128, 0, stream>>>(
      obuf, proj_w, proj_g, proj_b, proj_m, proj_v, out);
}
